// ReversibleSwinTransformerBlock_48395691491857
// MI455X (gfx1250) — matrix operand order repaired and checked by reading, not compiled, not run on hardware
//
#include <hip/hip_runtime.h>
#include <hip/hip_bf16.h>

typedef _Float16 half_t;
typedef __attribute__((ext_vector_type(16))) _Float16 v16h;
typedef __attribute__((ext_vector_type(8)))  _Float16 v8h;
typedef __attribute__((ext_vector_type(8)))  float    v8f;
typedef __attribute__((ext_vector_type(4)))  unsigned u32x4;
typedef __attribute__((ext_vector_type(8)))  int      i32x8;
typedef __attribute__((ext_vector_type(4)))  int      i32x4;

#define DIM    384
#define HEADS  12
#define BB     16
#define NWIN   64
#define ROWS   (BB * 56 * 56)        /* 50176 */
#define ROWS_P (ROWS + 16)           /* padded rows for window tail reads */
#define QKVN   1152
#define FCN    1536
#define LDS_STRIDE 40                /* 32 halves + 8 pad halves (4 dwords) */

#if __has_builtin(__builtin_amdgcn_tensor_load_to_lds)
#define HAVE_TDM 1
#else
#define HAVE_TDM 0
#endif

// ---------------------------------------------------------------------------
// Single-instruction XOR lane exchange via ds_swizzle_b32 (group-of-32 mode:
// offset[14:10]=xor_mask, offset[9:5]=or_mask=0, offset[4:0]=and_mask=0x1F).
// All lanes are active wherever these are used.
// ---------------------------------------------------------------------------
template <int XORMASK>
__device__ inline float swz_xor(float v) {
    int i = __builtin_bit_cast(int, v);
    i = __builtin_amdgcn_ds_swizzle(i, (XORMASK << 10) | 0x1F);
    return __builtin_bit_cast(float, i);
}

__device__ inline float wave_sum32(float v) {
    v += swz_xor<16>(v); v += swz_xor<8>(v);
    v += swz_xor<4>(v);  v += swz_xor<2>(v); v += swz_xor<1>(v);
    return v;
}
__device__ inline float half_sum16(float v) {   // reduce within 16-lane halves
    v += swz_xor<8>(v); v += swz_xor<4>(v);
    v += swz_xor<2>(v); v += swz_xor<1>(v);
    return v;
}
__device__ inline float half_max16(float v) {
    v = fmaxf(v, swz_xor<8>(v)); v = fmaxf(v, swz_xor<4>(v));
    v = fmaxf(v, swz_xor<2>(v)); v = fmaxf(v, swz_xor<1>(v));
    return v;
}

// ---------------------------------------------------------------------------
// WMMA fragment helpers (layouts per CDNA5 ISA 7.12.2, wave32)
// ---------------------------------------------------------------------------
__device__ inline v16h make_v16(v8h lo, v8h hi) {
    v16h r;
#pragma unroll
    for (int i = 0; i < 8; ++i) { r[i] = lo[i]; r[i + 8] = hi[i]; }
    return r;
}

// A fragment from padded LDS tile (row stride LDS_STRIDE halves).
__device__ inline v16h load_a_frag_lds(const half_t* Ab, int mrel, int lane) {
    int l    = lane & 15;
    int koff = (lane & 16) ? 16 : 0;
    const half_t* p = Ab + (mrel + l) * LDS_STRIDE + koff;
    return make_v16(*(const v8h*)p, *(const v8h*)(p + 8));
}

// B fragment: 32x16 f16 from K-major (transposed) weights Wt[N][K].
__device__ inline v16h load_b_frag(const half_t* Wt, int ldk, int n0, int k0, int lane) {
    int n  = n0 + (lane & 15);
    int kb = k0 + ((lane & 16) ? 16 : 0);
    return *(const v16h*)(Wt + (size_t)n * ldk + kb);
}

__device__ inline v8f wmma_f16(v16h a, v16h b, v8f c) {
    return __builtin_amdgcn_wmma_f32_16x16x32_f16(false, a, false, b, (short)0, c, false, false);
}

// ---------------------------------------------------------------------------
// Tensor Data Mover: stage a 128-row x 32-half A tile (row stride K halves)
// into LDS with +4-dword padding every 16 dwords (row stride 40 halves).
// ---------------------------------------------------------------------------
#if HAVE_TDM
__device__ inline void tdm_load_a(unsigned lds_off, const half_t* gptr, int K) {
    unsigned long long ga = (unsigned long long)(uintptr_t)gptr;
    u32x4 g0;
    g0[0] = 1u;                                        // count=1, user mode
    g0[1] = lds_off;                                   // LDS byte address
    g0[2] = (unsigned)ga;                              // global addr lo
    g0[3] = (unsigned)((ga >> 32) & 0x01FFFFFFu)       // global addr hi (bits 56:32)
          | 0x80000000u;                               // type=2 (bits 127:126 = 10b)
    i32x8 g1;
    g1[0] = (1 << 16)          // data_size = 2 bytes
          | (1 << 20)          // pad_enable
          | (3 << 22)          // pad_interval: every 16 dwords
          | (3 << 25);         // pad_amount: 4 dwords
    g1[1] = (K & 0xFFFF) << 16;   // tensor_dim0[15:0] = K
    g1[2] = (128 << 16);          // tensor_dim1[15:0] = 128 rows
    g1[3] = (32 << 16);           // tile_dim0 = 32 elements
    g1[4] = 128;                  // tile_dim1 = 128, tile_dim2 = 0
    g1[5] = K;                    // tensor_dim0_stride lo32 = K
    g1[6] = 0;
    g1[7] = 0;
    i32x4 z4 = (i32x4){0, 0, 0, 0};
    i32x8 z8 = (i32x8){0, 0, 0, 0, 0, 0, 0, 0};
    __builtin_amdgcn_tensor_load_to_lds(g0, g1, z4, z4, z8, 0);
}
#endif

// ---------------------------------------------------------------------------
// Weight prep: f32 [K][Nc] row-major -> f16 [Nc][K] (K-major per output col)
// ---------------------------------------------------------------------------
__global__ __launch_bounds__(256) void wprep_kernel(const float* __restrict__ w,
                                                    half_t* __restrict__ wt,
                                                    int K, int Nc) {
    int idx = blockIdx.x * 256 + threadIdx.x;
    if (idx >= K * Nc) return;
    int n = idx / K;
    int k = idx - n * K;
    wt[idx] = (half_t)w[(size_t)k * Nc + n];
}

// ---------------------------------------------------------------------------
// Combined relative-position bias + shift mask table:
// cmb[wi][head][nq][64], padded cols (nk>=49) pre-set to -1e30.
// ---------------------------------------------------------------------------
__global__ __launch_bounds__(256) void bias_kernel(const float* __restrict__ rpb,
                                                   const float* __restrict__ amask,
                                                   float* __restrict__ cmb) {
    int idx = blockIdx.x * 256 + threadIdx.x;
    if (idx >= NWIN * HEADS * 49 * 64) return;
    int nk = idx & 63;
    int t  = idx >> 6;
    int nq = t % 49; t /= 49;
    int h  = t % HEADS;
    int wi = t / HEADS;
    float v = -1e30f;
    if (nk < 49) {
        int iq = nq / 7, jq = nq - iq * 7;
        int ik = nk / 7, jk = nk - ik * 7;
        v = rpb[((iq - ik + 6) * 13 + (jq - jk + 6)) * HEADS + h]
          + amask[(size_t)wi * 2401 + nq * 49 + nk];
    }
    cmb[idx] = v;
}

// ---------------------------------------------------------------------------
// LayerNorm: one wave per row of 384 channels.
// mode 0: output in shifted+window-partitioned order ; mode 1: natural order
// ---------------------------------------------------------------------------
__global__ __launch_bounds__(256) void ln_kernel(const float* __restrict__ x,
                                                 const float* __restrict__ g,
                                                 const float* __restrict__ b,
                                                 half_t* __restrict__ out,
                                                 int mode) {
    int lane = threadIdx.x & 31;
    int row  = blockIdx.x * 8 + (threadIdx.x >> 5);
    if (row >= ROWS) return;

    int src = row;
    if (mode == 0) {
        int win = row / 49, n = row - win * 49;
        int bi  = win >> 6, wi = win & 63;
        int h   = (wi >> 3) * 7 + n / 7;
        int w   = (wi & 7) * 7 + n % 7;
        int sh  = h + 3; if (sh >= 56) sh -= 56;   // roll(-3,-3)
        int sw  = w + 3; if (sw >= 56) sw -= 56;
        src = bi * 3136 + sh * 56 + sw;
    }
    const float* xp = x + (size_t)src * DIM;

    float v[12];
    float s = 0.f;
#pragma unroll
    for (int t = 0; t < 12; ++t) { v[t] = xp[lane + 32 * t]; s += v[t]; }
    s = wave_sum32(s);
    float mu = s * (1.0f / DIM);
    float s2 = 0.f;
#pragma unroll
    for (int t = 0; t < 12; ++t) { float d = v[t] - mu; s2 += d * d; }
    s2 = wave_sum32(s2);
    float rs = rsqrtf(s2 * (1.0f / DIM) + 1e-5f);

    half_t* op = out + (size_t)row * DIM;
#pragma unroll
    for (int t = 0; t < 12; ++t) {
        int c = lane + 32 * t;
        op[c] = (half_t)((v[t] - mu) * rs * g[c] + b[c]);
    }
}

// ---------------------------------------------------------------------------
// WMMA GEMM with TDM-staged, double-buffered A tiles in LDS.
// Block tile 128x128, 8 waves, each wave: 32x64 via 2x4 16x16 accumulators.
// EP: 0 = QKV (scale q cols, f16 out)   1 = PROJ (reverse+shift+residual f32)
//     2 = FC1 (exact GELU, f16 out)     3 = FC2 (residual natural f32)
// ---------------------------------------------------------------------------
template <int EP>
__global__ __launch_bounds__(256)
void gemm_kernel(const half_t* __restrict__ A, const half_t* __restrict__ Wt,
                 const float* __restrict__ bias, int M, int N, int K,
                 half_t* __restrict__ outH, float* __restrict__ outF,
                 const float* __restrict__ addF) {
    __shared__ __align__(16) half_t Abuf[2][128 * LDS_STRIDE];

    int lane = threadIdx.x & 31;
    int wave = threadIdx.x >> 5;
    int wm   = wave & 3;
    int wn   = wave >> 2;
    int m0   = blockIdx.y * 128 + wm * 32;
    int n0   = blockIdx.x * 128 + wn * 64;
    const half_t* Ablk = A + (size_t)(blockIdx.y * 128) * K;

    v8f acc[2][4];
#pragma unroll
    for (int i = 0; i < 2; ++i)
#pragma unroll
        for (int j = 0; j < 4; ++j) acc[i][j] = (v8f){0.f,0.f,0.f,0.f,0.f,0.f,0.f,0.f};

    int nks = K >> 5;

#if HAVE_TDM
    if (wave == 0 && lane == 0)
        tdm_load_a((unsigned)(uintptr_t)&Abuf[0][0], Ablk, K);
    for (int ks = 0; ks < nks; ++ks) {
        if (wave == 0 && lane == 0) {
            if (ks + 1 < nks) {
                tdm_load_a((unsigned)(uintptr_t)&Abuf[(ks + 1) & 1][0],
                           Ablk + (ks + 1) * 32, K);
                __builtin_amdgcn_s_wait_tensorcnt(1);
            } else {
                __builtin_amdgcn_s_wait_tensorcnt(0);
            }
        }
        __syncthreads();
        const half_t* Ab = &Abuf[ks & 1][0];
        int k0 = ks * 32;
        v16h a0 = load_a_frag_lds(Ab, wm * 32,      lane);
        v16h a1 = load_a_frag_lds(Ab, wm * 32 + 16, lane);
        v16h bfr[4];
#pragma unroll
        for (int j = 0; j < 4; ++j) bfr[j] = load_b_frag(Wt, K, n0 + 16 * j, k0, lane);
#pragma unroll
        for (int j = 0; j < 4; ++j) {
            acc[0][j] = wmma_f16(a0, bfr[j], acc[0][j]);
            acc[1][j] = wmma_f16(a1, bfr[j], acc[1][j]);
        }
        __syncthreads();
    }
#else
    for (int ks = 0; ks < nks; ++ks) {
        int k0 = ks * 32;
        // cooperative fallback copy (16 dwords per row)
        for (int idx = threadIdx.x; idx < 128 * 16; idx += 256) {
            int rr = idx >> 4, dw = idx & 15;
            ((unsigned*)&Abuf[0][0])[rr * (LDS_STRIDE / 2) + dw] =
                ((const unsigned*)(Ablk + (size_t)rr * K + k0))[dw];
        }
        __syncthreads();
        const half_t* Ab = &Abuf[0][0];
        v16h a0 = load_a_frag_lds(Ab, wm * 32,      lane);
        v16h a1 = load_a_frag_lds(Ab, wm * 32 + 16, lane);
        v16h bfr[4];
#pragma unroll
        for (int j = 0; j < 4; ++j) bfr[j] = load_b_frag(Wt, K, n0 + 16 * j, k0, lane);
#pragma unroll
        for (int j = 0; j < 4; ++j) {
            acc[0][j] = wmma_f16(a0, bfr[j], acc[0][j]);
            acc[1][j] = wmma_f16(a1, bfr[j], acc[1][j]);
        }
        __syncthreads();
    }
#endif

    int colb   = lane & 15;
    int rowadd = (lane & 16) ? 8 : 0;
    const float SCALE = 0.17677669529663687f; // 1/sqrt(32)

#pragma unroll
    for (int i = 0; i < 2; ++i) {
#pragma unroll
        for (int r = 0; r < 8; ++r) {
            int row = m0 + i * 16 + r + rowadd;
            if (EP == 1) {
                int win = row / 49, n = row - win * 49;
                int bi  = win >> 6, wi = win & 63;
                int h   = (wi >> 3) * 7 + n / 7;
                int w   = (wi & 7) * 7 + n % 7;
                int oh  = h + 3; if (oh >= 56) oh -= 56;  // roll(+3,+3)
                int ow  = w + 3; if (ow >= 56) ow -= 56;
                size_t ob = ((size_t)bi * 3136 + oh * 56 + ow) * DIM;
#pragma unroll
                for (int j = 0; j < 4; ++j) {
                    int col = n0 + j * 16 + colb;
                    float v = acc[i][j][r] + bias[col];
                    outF[ob + col] = addF[ob + col] + v;
                }
            } else if (EP == 3) {
                size_t ob = (size_t)row * DIM;
#pragma unroll
                for (int j = 0; j < 4; ++j) {
                    int col = n0 + j * 16 + colb;
                    float v = acc[i][j][r] + bias[col];
                    outF[ob + col] = addF[ob + col] + v;
                }
            } else {
                half_t* op = outH + (size_t)row * N;
#pragma unroll
                for (int j = 0; j < 4; ++j) {
                    int col = n0 + j * 16 + colb;
                    float v = acc[i][j][r] + bias[col];
                    if (EP == 0) { if (col < DIM) v *= SCALE; }
                    if (EP == 2) { v = 0.5f * v * (1.0f + erff(v * 0.70710678118654752f)); }
                    op[col] = (half_t)v;
                }
            }
        }
    }
}

// ---------------------------------------------------------------------------
// Windowed attention: one block (128 thr = 4 waves) per (window, head).
// QK^T via WMMA, fused bias+mask table, softmax in accumulator layout,
// P and V^T staged in LDS, PV via WMMA.
// ---------------------------------------------------------------------------
__global__ __launch_bounds__(128)
void attn_kernel(const half_t* __restrict__ qkv, const float* __restrict__ cmb,
                 half_t* __restrict__ ao) {
    __shared__ __align__(32) half_t VT[32 * 64];       // V^T, zero padded cols
    __shared__ __align__(32) half_t Psh[4][16 * 64];   // per-wave prob tiles

    int win  = blockIdx.x;
    int head = blockIdx.y;
    int tid  = threadIdx.x;
    int lane = tid & 31;
    int wave = tid >> 5;
    size_t base = (size_t)win * 49;
    int wi = win & 63;
    const float* cb = cmb + ((size_t)wi * HEADS + head) * 49 * 64;

    // stage V^T (32 x 64), zero the n >= 49 padding
    for (int idx = tid; idx < 2048; idx += 128) {
        int c = idx >> 6, n = idx & 63;
        half_t v = (half_t)0;
        if (n < 49) v = qkv[(base + n) * QKVN + 2 * DIM + head * 32 + c];
        VT[idx] = v;
    }
    __syncthreads();

    int m0     = wave * 16;
    int colb   = lane & 15;
    int rowadd = (lane & 16) ? 8 : 0;

    // ---- scores = q @ k^T  (q pre-scaled in QKV epilogue) ----
    v8f acc[4];
#pragma unroll
    for (int j = 0; j < 4; ++j) acc[j] = (v8f){0.f,0.f,0.f,0.f,0.f,0.f,0.f,0.f};
    {
        const half_t* qp = qkv + (base + m0 + colb) * QKVN + head * 32 + ((lane & 16) ? 16 : 0);
        v16h a = make_v16(*(const v8h*)qp, *(const v8h*)(qp + 8));
#pragma unroll
        for (int j = 0; j < 4; ++j) {
            int n = j * 16 + colb;
            const half_t* kp = qkv + (base + n) * QKVN + DIM + head * 32 + ((lane & 16) ? 16 : 0);
            v16h b = *(const v16h*)kp;
            acc[j] = wmma_f16(a, b, acc[j]);
        }
    }

    // ---- + combined bias/mask, then row softmax (16-lane-half reduce) ----
#pragma unroll
    for (int r = 0; r < 8; ++r) {
        int nq = m0 + r + rowadd;
        float x[4];
        if (nq < 49) {
            const float* cr = cb + nq * 64;
#pragma unroll
            for (int j = 0; j < 4; ++j) x[j] = acc[j][r] + cr[j * 16 + colb];
        } else {
#pragma unroll
            for (int j = 0; j < 4; ++j) x[j] = -1e30f;
        }
        float mx = half_max16(fmaxf(fmaxf(x[0], x[1]), fmaxf(x[2], x[3])));
        float e0 = __expf(x[0] - mx), e1 = __expf(x[1] - mx);
        float e2 = __expf(x[2] - mx), e3 = __expf(x[3] - mx);
        float s = half_sum16(e0 + e1 + e2 + e3);
        float is = 1.0f / s;
        int prow = (r + rowadd) * 64;
        Psh[wave][prow + 0 * 16 + colb] = (half_t)(e0 * is);
        Psh[wave][prow + 1 * 16 + colb] = (half_t)(e1 * is);
        Psh[wave][prow + 2 * 16 + colb] = (half_t)(e2 * is);
        Psh[wave][prow + 3 * 16 + colb] = (half_t)(e3 * is);
    }

    // ---- out = P @ V  (K = 64 padded, N = 32) ----
    v8f oacc[2];
    oacc[0] = (v8f){0.f,0.f,0.f,0.f,0.f,0.f,0.f,0.f};
    oacc[1] = (v8f){0.f,0.f,0.f,0.f,0.f,0.f,0.f,0.f};
#pragma unroll
    for (int kt = 0; kt < 2; ++kt) {
        int c0 = kt * 32 + ((lane & 16) ? 16 : 0);
        const half_t* pp = &Psh[wave][colb * 64 + c0];
        v16h a = make_v16(*(const v8h*)pp, *(const v8h*)(pp + 8));
#pragma unroll
        for (int ct = 0; ct < 2; ++ct) {
            int c  = ct * 16 + colb;
            int kb = kt * 32 + ((lane & 16) ? 16 : 0);
            v16h b = *(const v16h*)&VT[c * 64 + kb];
            oacc[ct] = wmma_f16(a, b, oacc[ct]);
        }
    }

#pragma unroll
    for (int ct = 0; ct < 2; ++ct) {
#pragma unroll
        for (int r = 0; r < 8; ++r) {
            int nq = m0 + r + rowadd;
            if (nq < 49)
                ao[(base + nq) * DIM + head * 32 + ct * 16 + colb] = (half_t)oacc[ct][r];
        }
    }
}

// ---------------------------------------------------------------------------
// Host-side orchestration
// ---------------------------------------------------------------------------
extern "C" void kernel_launch(void* const* d_in, const int* in_sizes, int n_in,
                              void* d_out, int out_size, void* d_ws, size_t ws_size,
                              hipStream_t stream) {
    (void)in_sizes; (void)n_in; (void)out_size; (void)ws_size;

    const float* x1     = (const float*)d_in[0];
    const float* x2     = (const float*)d_in[1];
    const float* amask  = (const float*)d_in[2];
    const float* g1     = (const float*)d_in[3];
    const float* be1    = (const float*)d_in[4];
    const float* w_qkv  = (const float*)d_in[5];
    const float* b_qkv  = (const float*)d_in[6];
    const float* rpb    = (const float*)d_in[7];
    const float* w_proj = (const float*)d_in[8];
    const float* b_proj = (const float*)d_in[9];
    const float* g2     = (const float*)d_in[10];
    const float* be2    = (const float*)d_in[11];
    const float* w_fc1  = (const float*)d_in[12];
    const float* b_fc1  = (const float*)d_in[13];
    const float* w_fc2  = (const float*)d_in[14];
    const float* b_fc2  = (const float*)d_in[15];

    float* y1 = (float*)d_out;
    float* y2 = y1 + (size_t)ROWS * DIM;

    // workspace layout (halves)
    half_t* ws     = (half_t*)d_ws;
    half_t* wqkvT  = ws;                                   // [1152][384]
    half_t* wprojT = wqkvT  + (size_t)QKVN * DIM;          // [384][384]
    half_t* wfc1T  = wprojT + (size_t)DIM * DIM;           // [1536][384]
    half_t* wfc2T  = wfc1T  + (size_t)FCN * DIM;           // [384][1536]
    half_t* xw     = wfc2T  + (size_t)DIM * FCN;           // [ROWS_P][384] (reused as attn_out)
    half_t* qkv    = xw     + (size_t)ROWS_P * DIM;        // [ROWS_P][1152] (reused as LN2 out)
    half_t* h1     = qkv    + (size_t)ROWS_P * QKVN;       // [ROWS][1536]
    float*  cmb    = (float*)(h1 + (size_t)ROWS * FCN);    // [64][12][49][64]
    half_t* ao     = xw;
    half_t* h1in   = qkv;

    // 1. weight transposes + fused bias/mask table
    wprep_kernel<<<(DIM * QKVN + 255) / 256, 256, 0, stream>>>(w_qkv,  wqkvT, DIM, QKVN);
    wprep_kernel<<<(DIM * DIM  + 255) / 256, 256, 0, stream>>>(w_proj, wprojT, DIM, DIM);
    wprep_kernel<<<(DIM * FCN  + 255) / 256, 256, 0, stream>>>(w_fc1,  wfc1T, DIM, FCN);
    wprep_kernel<<<(FCN * DIM  + 255) / 256, 256, 0, stream>>>(w_fc2,  wfc2T, FCN, DIM);
    bias_kernel<<<(NWIN * HEADS * 49 * 64 + 255) / 256, 256, 0, stream>>>(rpb, amask, cmb);

    // 2. LN1 + shift + window partition: x2 -> xw (f16)
    ln_kernel<<<ROWS / 8, 256, 0, stream>>>(x2, g1, be1, xw, 0);

    // 3. QKV projection (q scaled by 1/sqrt(head_dim))
    gemm_kernel<0><<<dim3(QKVN / 128, ROWS / 128), 256, 0, stream>>>(
        xw, wqkvT, b_qkv, ROWS, QKVN, DIM, qkv, nullptr, nullptr);

    // 4. windowed multi-head attention -> ao (f16)
    attn_kernel<<<dim3(BB * NWIN, HEADS), 128, 0, stream>>>(qkv, cmb, ao);

    // 5. output projection + window reverse + unshift + residual: y1 = x1 + ...
    gemm_kernel<1><<<dim3(DIM / 128, ROWS / 128), 256, 0, stream>>>(
        ao, wprojT, b_proj, ROWS, DIM, DIM, nullptr, y1, x1);

    // 6. LN2 on y1 -> h1in (f16)
    ln_kernel<<<ROWS / 8, 256, 0, stream>>>(y1, g2, be2, h1in, 1);

    // 7. FC1 + exact GELU
    gemm_kernel<2><<<dim3(FCN / 128, ROWS / 128), 256, 0, stream>>>(
        h1in, wfc1T, b_fc1, ROWS, FCN, DIM, h1, nullptr, nullptr);

    // 8. FC2 + residual: y2 = x2 + ...
    gemm_kernel<3><<<dim3(DIM / 128, ROWS / 128), 256, 0, stream>>>(
        h1, wfc2T, b_fc2, ROWS, DIM, FCN, nullptr, y2, x2);
}
